// Memorynet_66048007078358
// MI455X (gfx1250) — matrix operand order repaired and checked by reading, not compiled, not run on hardware
//
#include <hip/hip_runtime.h>
#include <hip/hip_bf16.h>

// ---------------------------------------------------------------------------
// MI455X (gfx1250) fused MemoryNet feature-propagation block.
//   K1: fold BN params, convert weights to bf16  (tiny)
//   K2: KNN(k=3) inverse-distance interpolation + concat -> X bf16 (M x 384)
//   K3: fused X@W1^T -> BN/ReLU -> (LDS) -> @W2^T -> BN/ReLU -> out
// GEMMs: v_wmma_f32_16x16x32_bf16 (wave32, fp32 accumulate).
// X tile staged into LDS with the Tensor Data Mover when available.
// ---------------------------------------------------------------------------

typedef __attribute__((ext_vector_type(16))) __bf16        v16bf;
typedef __attribute__((ext_vector_type(8)))  float         v8f;
typedef __attribute__((ext_vector_type(4)))  unsigned int  u32x4;
typedef __attribute__((ext_vector_type(8)))  int           i32x8;
typedef __attribute__((ext_vector_type(4)))  int           i32x4;

#define B_   32
#define N1_  2048
#define N2_  512
#define C1_  128
#define C2_  256
#define CIN_ 384   // C1_ + C2_
#define H1_  256
#define H2_  128
#define M_   (B_ * N1_)   // 65536 rows

#if __has_builtin(__builtin_amdgcn_tensor_load_to_lds) && \
    __has_builtin(__builtin_amdgcn_s_wait_tensorcnt)
#define USE_TDM 1
#else
#define USE_TDM 0
#endif

__device__ __forceinline__ unsigned short f32_to_bf16(float f) {
    unsigned int u = __float_as_uint(f);
    u += 0x7FFFu + ((u >> 16) & 1u);          // round-to-nearest-even
    return (unsigned short)(u >> 16);
}
__device__ __forceinline__ unsigned int pack_bf16x2(float a, float b) {
    return (unsigned int)f32_to_bf16(a) | ((unsigned int)f32_to_bf16(b) << 16);
}

// ---------------------------------------------------------------------------
// K1: weight conversion + BN folding.  out = max(0, y*scale + shift)
//     scale = g/sqrt(v+eps) ; shift = (b - m)*scale + be
// ---------------------------------------------------------------------------
__global__ void prep_kernel(const float* __restrict__ W1, const float* __restrict__ W2,
                            const float* __restrict__ b1, const float* __restrict__ g1,
                            const float* __restrict__ be1, const float* __restrict__ m1,
                            const float* __restrict__ v1,
                            const float* __restrict__ b2, const float* __restrict__ g2,
                            const float* __restrict__ be2, const float* __restrict__ m2,
                            const float* __restrict__ v2,
                            unsigned short* __restrict__ W1bf, unsigned short* __restrict__ W2bf,
                            float* __restrict__ sc1, float* __restrict__ sh1,
                            float* __restrict__ sc2, float* __restrict__ sh2) {
    const int i = blockIdx.x * blockDim.x + threadIdx.x;
    if (i < H1_ * CIN_) W1bf[i] = f32_to_bf16(W1[i]);
    if (i < H2_ * H1_)  W2bf[i] = f32_to_bf16(W2[i]);
    if (i < H1_) {
        float s = g1[i] * rsqrtf(v1[i] + 1e-5f);
        sc1[i] = s; sh1[i] = (b1[i] - m1[i]) * s + be1[i];
    }
    if (i < H2_) {
        float s = g2[i] * rsqrtf(v2[i] + 1e-5f);
        sc2[i] = s; sh2[i] = (b2[i] - m2[i]) * s + be2[i];
    }
}

// ---------------------------------------------------------------------------
// K2: per-query top-3 scan over 512 LDS-staged points, inverse-distance blend
//     of features_2 (L2-resident), write X = [recv | features_1] as bf16.
// ---------------------------------------------------------------------------
__global__ __launch_bounds__(256)
void knn_concat_kernel(const float* __restrict__ points_1,
                       const float* __restrict__ features_1,
                       const float* __restrict__ points_2,
                       const float* __restrict__ features_2,
                       unsigned short* __restrict__ X) {
    __shared__ float p2s[N2_ * 3];                       // 6 KB
    const int bb = blockIdx.y;
    const int q  = blockIdx.x * blockDim.x + threadIdx.x;   // 0..2047

    const float* p2g = points_2 + (size_t)bb * N2_ * 3;
    for (int i = threadIdx.x; i < N2_ * 3; i += blockDim.x) p2s[i] = p2g[i];
    __syncthreads();

    const float* pq = points_1 + ((size_t)bb * N1_ + q) * 3;
    const float px = pq[0], py = pq[1], pz = pq[2];

    float d0 = 3.4e38f, d1 = 3.4e38f, d2 = 3.4e38f;
    int   i0 = 0, i1 = 0, i2 = 0;
    #pragma unroll 4
    for (int j = 0; j < N2_; ++j) {
        float dx = px - p2s[j * 3 + 0];
        float dy = py - p2s[j * 3 + 1];
        float dz = pz - p2s[j * 3 + 2];
        float d  = dx * dx + dy * dy + dz * dz;
        if (d < d0)       { d2 = d1; i2 = i1; d1 = d0; i1 = i0; d0 = d; i0 = j; }
        else if (d < d1)  { d2 = d1; i2 = i1; d1 = d;  i1 = j; }
        else if (d < d2)  { d2 = d;  i2 = j; }
    }
    float r0 = 1.0f / (d0 + 1e-8f);
    float r1 = 1.0f / (d1 + 1e-8f);
    float r2 = 1.0f / (d2 + 1e-8f);
    float rs = 1.0f / (r0 + r1 + r2);
    const float w0 = r0 * rs, w1 = r1 * rs, w2 = r2 * rs;

    const float4* f0 = (const float4*)(features_2 + ((size_t)bb * N2_ + i0) * C2_);
    const float4* f1 = (const float4*)(features_2 + ((size_t)bb * N2_ + i1) * C2_);
    const float4* f2 = (const float4*)(features_2 + ((size_t)bb * N2_ + i2) * C2_);
    unsigned short* xr = X + ((size_t)bb * N1_ + q) * CIN_;

    #pragma unroll 4
    for (int c4 = 0; c4 < C2_ / 4; ++c4) {               // interpolated features
        float4 a = f0[c4], b = f1[c4], c = f2[c4];
        uint2 o;
        o.x = pack_bf16x2(w0 * a.x + w1 * b.x + w2 * c.x,
                          w0 * a.y + w1 * b.y + w2 * c.y);
        o.y = pack_bf16x2(w0 * a.z + w1 * b.z + w2 * c.z,
                          w0 * a.w + w1 * b.w + w2 * c.w);
        *(uint2*)(xr + 4 * c4) = o;
    }
    const float4* ff = (const float4*)(features_1 + ((size_t)bb * N1_ + q) * C1_);
    unsigned short* xr2 = xr + C2_;
    #pragma unroll 4
    for (int c4 = 0; c4 < C1_ / 4; ++c4) {               // skip-connected features
        float4 a = ff[c4];
        uint2 o;
        o.x = pack_bf16x2(a.x, a.y);
        o.y = pack_bf16x2(a.z, a.w);
        *(uint2*)(xr2 + 4 * c4) = o;
    }
}

// ---------------------------------------------------------------------------
// WMMA fragment loaders (ISA 7.12.2, wave32, 16-bit operands).
// A (16x32): lane = {m 0..15} x {half}; VGPR0..3 <- K = 8h..8h+7,
//            VGPR4..7 <- K = 16+8h..16+8h+7  => two b128 loads.
// B (KxN slice as W row-major NxK): lane n = lane&15; VGPR0..3 <- K = 8h..8h+7,
//            VGPR4..7 <- K = 16+8h..16+8h+7  => two b128 loads along K.
// C/D (16x16 f32): lane n = lane&15; VGPRv <- M = (lane>>4)*8 + v.
// ---------------------------------------------------------------------------
__device__ __forceinline__ v16bf load_a_frag(const unsigned short* base, int ldk,
                                             int lane, int kb) {
    const int m = lane & 15, h = lane >> 4;
    const unsigned short* p = base + m * ldk + kb + 8 * h;
    union { v16bf v; uint4 q[2]; } u;
    u.q[0] = *(const uint4*)(p);
    u.q[1] = *(const uint4*)(p + 16);
    return u.v;
}
__device__ __forceinline__ v16bf load_b_frag(const unsigned short* W, int ldk,
                                             int lane, int n0, int kb) {
    const int n = n0 + (lane & 15);
    const unsigned short* p = W + (size_t)n * ldk + kb + (lane >> 4) * 8;
    union { v16bf v; uint4 q[2]; } u;
    u.q[0] = *(const uint4*)(p);
    u.q[1] = *(const uint4*)(p + 16);
    return u.v;
}

#define TM 32   // rows per block

__global__ __launch_bounds__(128)
void fused_mlp_kernel(const unsigned short* __restrict__ X,     // M x 384 bf16
                      const unsigned short* __restrict__ W1bf,  // 256 x 384 bf16
                      const unsigned short* __restrict__ W2bf,  // 128 x 256 bf16
                      const float* __restrict__ sc1, const float* __restrict__ sh1,
                      const float* __restrict__ sc2, const float* __restrict__ sh2,
                      float* __restrict__ out) {                // M x 128 f32
    __shared__ unsigned short Xs[TM * CIN_];   // 24 KB
    __shared__ unsigned short Hs[TM * H1_];    // 16 KB

    const int tid  = threadIdx.x;
    const int lane = tid & 31;
    const int wid  = tid >> 5;                 // 4 waves
    const size_t m0 = (size_t)blockIdx.x * TM;

    // warm L2 path for the (resident) second-layer weights
    __builtin_prefetch((const void*)(W2bf + (size_t)tid * 256), 0, 1);

    // ---- stage X tile (32 x 384 bf16 = 24 KB) into LDS ----
#if USE_TDM
    if (wid == 0) {
        // Tensor DMA descriptor (ISA 8.3/8.4): 2-D tile, data_size=2B,
        // tile 384 x 32, tensor_dim0(_stride)=384, type=2 ("image").
        const unsigned long long gaddr = (unsigned long long)(const void*)(X + m0 * CIN_);
        const unsigned int ldsoff = (unsigned int)(unsigned long long)(void*)&Xs[0];
        u32x4 g0;
        g0.x = 1u;                                            // count=1, user mode
        g0.y = ldsoff;                                        // lds_addr
        g0.z = (unsigned int)gaddr;                           // global_addr[31:0]
        g0.w = (unsigned int)((gaddr >> 32) & 0x01FFFFFFull)  // global_addr[56:32]
               | (2u << 30);                                  // type=2
        i32x8 g1;
        g1[0] = (int)(1u << 16);                 // wg_mask=0, data_size=1 (2B)
        g1[1] = (int)(384u << 16);               // tensor_dim0[15:0] @ [63:48]
        g1[2] = 0;                               // tensor_dim0 hi, tensor_dim1 lo
        g1[3] = (int)(1u | (384u << 16));        // tensor_dim1 hi=1 (65536), tile_dim0=384
        g1[4] = TM;                              // tile_dim1=32, tile_dim2=0
        g1[5] = 384;                             // tensor_dim0_stride[31:0]
        g1[6] = 0;
        g1[7] = 0;
        i32x4 gz; gz[0] = 0; gz[1] = 0; gz[2] = 0; gz[3] = 0;
#if __clang_major__ >= 23
        i32x8 gz8; gz8[0]=0; gz8[1]=0; gz8[2]=0; gz8[3]=0; gz8[4]=0; gz8[5]=0; gz8[6]=0; gz8[7]=0;
        __builtin_amdgcn_tensor_load_to_lds(g0, g1, gz, gz, gz8, 0);
#else
        __builtin_amdgcn_tensor_load_to_lds(g0, g1, gz, gz, 0);
#endif
        __builtin_amdgcn_s_wait_tensorcnt(0);
    }
#else
    {
        const uint4* src = (const uint4*)(X + m0 * CIN_);
        uint4* dst = (uint4*)Xs;
        #pragma unroll
        for (int i = 0; i < (TM * CIN_) / (8 * 128); ++i)
            dst[tid + i * 128] = src[tid + i * 128];
    }
#endif
    __syncthreads();

    const int nlo = lane & 15;
    const int mb  = (lane >> 4) * 8;

    // ---- layer 1: 32(M) x 64(N) per wave; 4 waves cover N=256 ----
    {
        const int cb = wid * 64;
        v8f acc[2][4] = {};
        for (int kb = 0; kb < CIN_; kb += 32) {
            v16bf a0 = load_a_frag(Xs,             CIN_, lane, kb);
            v16bf a1 = load_a_frag(Xs + 16 * CIN_, CIN_, lane, kb);
            #pragma unroll
            for (int t = 0; t < 4; ++t) {
                v16bf b = load_b_frag(W1bf, CIN_, lane, cb + t * 16, kb);
                acc[0][t] = __builtin_amdgcn_wmma_f32_16x16x32_bf16(
                    false, a0, false, b, (short)0, acc[0][t], false, false);
                acc[1][t] = __builtin_amdgcn_wmma_f32_16x16x32_bf16(
                    false, a1, false, b, (short)0, acc[1][t], false, false);
            }
        }
        #pragma unroll
        for (int r = 0; r < 2; ++r) {
            #pragma unroll
            for (int t = 0; t < 4; ++t) {
                const int n = cb + t * 16 + nlo;
                const float s = sc1[n], c = sh1[n];
                #pragma unroll
                for (int v = 0; v < 8; ++v) {
                    float h = acc[r][t][v] * s + c;
                    h = h > 0.0f ? h : 0.0f;       // ReLU
                    Hs[(r * 16 + mb + v) * H1_ + n] = f32_to_bf16(h);
                }
            }
        }
    }
    __syncthreads();

    // ---- layer 2: 32(M) x 32(N) per wave; 4 waves cover N=128 ----
    {
        const int cb = wid * 32;
        v8f acc[2][2] = {};
        for (int kb = 0; kb < H1_; kb += 32) {
            v16bf a0 = load_a_frag(Hs,            H1_, lane, kb);
            v16bf a1 = load_a_frag(Hs + 16 * H1_, H1_, lane, kb);
            #pragma unroll
            for (int t = 0; t < 2; ++t) {
                v16bf b = load_b_frag(W2bf, H1_, lane, cb + t * 16, kb);
                acc[0][t] = __builtin_amdgcn_wmma_f32_16x16x32_bf16(
                    false, a0, false, b, (short)0, acc[0][t], false, false);
                acc[1][t] = __builtin_amdgcn_wmma_f32_16x16x32_bf16(
                    false, a1, false, b, (short)0, acc[1][t], false, false);
            }
        }
        #pragma unroll
        for (int r = 0; r < 2; ++r) {
            #pragma unroll
            for (int t = 0; t < 2; ++t) {
                const int n = cb + t * 16 + nlo;
                const float s = sc2[n], c = sh2[n];
                #pragma unroll
                for (int v = 0; v < 8; ++v) {
                    float h = acc[r][t][v] * s + c;
                    h = h > 0.0f ? h : 0.0f;
                    out[(m0 + r * 16 + mb + v) * H2_ + n] = h;
                }
            }
        }
    }
}

// ---------------------------------------------------------------------------
extern "C" void kernel_launch(void* const* d_in, const int* in_sizes, int n_in,
                              void* d_out, int out_size, void* d_ws, size_t ws_size,
                              hipStream_t stream) {
    const float* points_1   = (const float*)d_in[0];
    const float* features_1 = (const float*)d_in[1];
    const float* points_2   = (const float*)d_in[2];
    const float* features_2 = (const float*)d_in[3];
    const float* W1  = (const float*)d_in[4];
    const float* b1  = (const float*)d_in[5];
    const float* g1  = (const float*)d_in[6];
    const float* be1 = (const float*)d_in[7];
    const float* m1  = (const float*)d_in[8];
    const float* v1  = (const float*)d_in[9];
    const float* W2  = (const float*)d_in[10];
    const float* b2  = (const float*)d_in[11];
    const float* g2  = (const float*)d_in[12];
    const float* be2 = (const float*)d_in[13];
    const float* m2  = (const float*)d_in[14];
    const float* v2  = (const float*)d_in[15];
    float* out = (float*)d_out;

    // workspace carve-up (all 16B aligned)
    char* ws = (char*)d_ws;
    size_t off = 0;
    unsigned short* Xbf  = (unsigned short*)(ws + off); off += (size_t)M_ * CIN_ * 2;  // 48 MB
    unsigned short* W1bf = (unsigned short*)(ws + off); off += (size_t)H1_ * CIN_ * 2;
    unsigned short* W2bf = (unsigned short*)(ws + off); off += (size_t)H2_ * H1_ * 2;
    float* sc1 = (float*)(ws + off); off += H1_ * 4;
    float* sh1 = (float*)(ws + off); off += H1_ * 4;
    float* sc2 = (float*)(ws + off); off += H2_ * 4;
    float* sh2 = (float*)(ws + off); off += H2_ * 4;
    (void)ws_size; (void)in_sizes; (void)n_in; (void)out_size;

    prep_kernel<<<(H1_ * CIN_ + 255) / 256, 256, 0, stream>>>(
        W1, W2, b1, g1, be1, m1, v1, b2, g2, be2, m2, v2,
        W1bf, W2bf, sc1, sh1, sc2, sh2);

    knn_concat_kernel<<<dim3(N1_ / 256, B_), 256, 0, stream>>>(
        points_1, features_1, points_2, features_2, Xbf);

    fused_mlp_kernel<<<M_ / TM, 128, 0, stream>>>(
        Xbf, W1bf, W2bf, sc1, sh1, sc2, sh2, out);
}
